// Head_5909874999480
// MI455X (gfx1250) — matrix operand order repaired and checked by reading, not compiled, not run on hardware
//
#include <hip/hip_runtime.h>

// ---------------------------------------------------------------------------
// Causal single-head attention, flash-attention style, for gfx1250 (MI455X).
// B=8, T=2048, C=1024, H=64. fp32 in/out, f16 WMMA compute w/ f32 accum.
// ---------------------------------------------------------------------------

typedef __attribute__((ext_vector_type(16))) _Float16 v16h;
typedef __attribute__((ext_vector_type(8)))  _Float16 v8h;
typedef __attribute__((ext_vector_type(8)))  float    v8f;

#define DEVI __device__ __forceinline__

static constexpr int BB = 8;      // batch
static constexpr int TT = 2048;   // sequence
static constexpr int CC = 1024;   // embed
static constexpr int HH = 64;     // head size

// ---- WMMA fragment helpers (CDNA5 16-bit layouts, wave32) -----------------
// A 16x32 (f16): lane L holds row M=L%16; lanes 0-15: K = {0..7, 16..23};
//                lanes 16-31: K = {8..15, 24..31}.
DEVI v16h ld_a(const _Float16* row, int hi) {
  const int o = hi ? 8 : 0;
  union { v16h v; v8h h[2]; } u;
  u.h[0] = *(const v8h*)(row + o);
  u.h[1] = *(const v8h*)(row + o + 16);
  return u.v;
}
// B 32x16 (f16): lane L holds col N=L%16; lanes 0-15: K = {0..7, 16..23};
//                lanes 16-31: K = {8..15, 24..31}. 'col' points at 32 K-contiguous halfs.
DEVI v16h ld_b(const _Float16* col, int hi) {
  const int o = hi ? 8 : 0;
  union { v16h v; v8h h[2]; } u;
  u.h[0] = *(const v8h*)(col + o);
  u.h[1] = *(const v8h*)(col + o + 16);
  return u.v;
}
DEVI v8f wmma_f16(v16h a, v16h b, v8f c) {
  // emits v_wmma_f32_16x16x32_f16 (codegen-confirmed builtin, 8 args)
  return __builtin_amdgcn_wmma_f32_16x16x32_f16(
      /*neg_a=*/false, a, /*neg_b=*/false, b,
      /*c_mod=*/(short)0, c, /*reuse_a=*/false, /*reuse_b=*/false);
}

// ---------------------------------------------------------------------------
// Kernel 0: transpose+convert weights  w[c][n] (f32) -> wT[mat][n][c] (f16)
// mat: 0=q, 1=k, 2=v
// ---------------------------------------------------------------------------
__global__ __launch_bounds__(256) void wt_kernel(const float* __restrict__ wq,
                                                 const float* __restrict__ wk,
                                                 const float* __restrict__ wv,
                                                 _Float16* __restrict__ wT) {
  const int mat = blockIdx.y;
  const float* w = (mat == 0) ? wq : (mat == 1) ? wk : wv;
  _Float16* o = wT + (size_t)mat * HH * CC;
  const int t = blockIdx.x * 256 + threadIdx.x;   // 0..65535
  const int c = t >> 6;
  const int n = t & 63;
  o[(size_t)n * CC + c] = (_Float16)w[t];
}

// ---------------------------------------------------------------------------
// Kernel 1: QKV projection.  out[mat][row][n] = sum_c x[row][c]*w[mat][c][n]
// grid = (B*T/128, 3), block = 256 (8 waves); wave w owns rows 16w..16w+15.
// ---------------------------------------------------------------------------
__global__ __launch_bounds__(256) void proj_kernel(const float* __restrict__ x,
                                                   const _Float16* __restrict__ wT,
                                                   _Float16* __restrict__ outq,
                                                   _Float16* __restrict__ outk,
                                                   _Float16* __restrict__ outv) {
  __shared__ alignas(16) _Float16 xs[128 * 40];   // 128 rows x 32 K, stride 40

  const int mat = blockIdx.y;
  const _Float16* wTm = wT + (size_t)mat * HH * CC;
  _Float16* out = (mat == 0) ? outq : (mat == 1) ? outk : outv;

  const int row0 = blockIdx.x * 128;
  const int tid  = threadIdx.x;
  const int lane = tid & 31;
  const int wv_  = tid >> 5;       // wave id 0..7
  const int hi   = lane >> 4;      // lane half
  const int ln   = lane & 15;

  const v8f vzero = {0.f, 0.f, 0.f, 0.f, 0.f, 0.f, 0.f, 0.f};
  v8f acc[4];
#pragma unroll
  for (int i = 0; i < 4; ++i) acc[i] = vzero;

  const int r  = tid >> 1;             // row this thread stages
  const int cs = (tid & 1) * 16;       // 16-col slab

  for (int c0 = 0; c0 < CC; c0 += 32) {
    // Stage x[row0+r][c0+cs .. +15] as f16 into LDS.
    const float* xp = x + (size_t)(row0 + r) * CC + c0 + cs;
    float4 a0 = ((const float4*)xp)[0];
    float4 a1 = ((const float4*)xp)[1];
    float4 a2 = ((const float4*)xp)[2];
    float4 a3 = ((const float4*)xp)[3];
    v8h h0 = {(_Float16)a0.x, (_Float16)a0.y, (_Float16)a0.z, (_Float16)a0.w,
              (_Float16)a1.x, (_Float16)a1.y, (_Float16)a1.z, (_Float16)a1.w};
    v8h h1 = {(_Float16)a2.x, (_Float16)a2.y, (_Float16)a2.z, (_Float16)a2.w,
              (_Float16)a3.x, (_Float16)a3.y, (_Float16)a3.z, (_Float16)a3.w};
    *(v8h*)&xs[r * 40 + cs]     = h0;
    *(v8h*)&xs[r * 40 + cs + 8] = h1;
    __syncthreads();

    const v16h af = ld_a(&xs[(wv_ * 16 + ln) * 40], hi);
#pragma unroll
    for (int nt = 0; nt < 4; ++nt) {
      const _Float16* bp = wTm + (size_t)(nt * 16 + ln) * CC + c0;  // L2-hot
      acc[nt] = wmma_f16(af, ld_b(bp, hi), acc[nt]);
    }
    __syncthreads();
  }

  // C layout: VGPR j -> row j (lanes 0-15) / row j+8 (lanes 16-31), col = ln.
#pragma unroll
  for (int nt = 0; nt < 4; ++nt)
#pragma unroll
    for (int j = 0; j < 8; ++j)
      out[(size_t)(row0 + wv_ * 16 + j + hi * 8) * HH + nt * 16 + ln] =
          (_Float16)acc[nt][j];
}

// ---------------------------------------------------------------------------
// Kernel 2: fused causal flash attention.
// grid = (T/128, B), block = 256 (8 waves). Wave w owns 16 query rows.
// ---------------------------------------------------------------------------
__global__ __launch_bounds__(256) void attn_kernel(const _Float16* __restrict__ qf,
                                                   const _Float16* __restrict__ kf,
                                                   const _Float16* __restrict__ vf,
                                                   float* __restrict__ out) {
  __shared__ alignas(16) _Float16 ks[128 * 72];      // K tile, stride 72
  __shared__ alignas(16) _Float16 vts[64 * 136];     // V^T tile, stride 136
  __shared__ alignas(16) _Float16 stage[8 * 16 * 40];// per-wave P staging

  const int b   = blockIdx.y;
  const int qt  = blockIdx.x;
  const int tid = threadIdx.x;
  const int lane = tid & 31;
  const int wv_  = tid >> 5;
  const int hi   = lane >> 4;
  const int ln   = lane & 15;

  const int qrow0 = qt * 128 + wv_ * 16;   // first query row of this wave

  // Q A-fragments live in registers for the whole kernel (2 chunks of K=32).
  const _Float16* qbase = qf + ((size_t)b * TT + qrow0 + ln) * HH;
  const v16h qa0 = ld_a(qbase, hi);
  const v16h qa1 = ld_a(qbase + 32, hi);

  const v8f vzero = {0.f, 0.f, 0.f, 0.f, 0.f, 0.f, 0.f, 0.f};
  v8f o[4];
#pragma unroll
  for (int i = 0; i < 4; ++i) o[i] = vzero;
  float m[8], l[8];
#pragma unroll
  for (int j = 0; j < 8; ++j) { m[j] = -1e30f; l[j] = 0.f; }

  const int r  = tid >> 1;
  const int cs = (tid & 1) * 32;
  const float scale = 0.03125f;          // C^-0.5 = 1/32

  for (int kt = 0; kt <= qt; ++kt) {
    // ---- cooperative stage: K -> ks (row major), V -> vts (transposed) ----
    {
      const size_t srow = (size_t)b * TT + kt * 128 + r;
      const _Float16* kp = kf + srow * HH + cs;
      const _Float16* vp = vf + srow * HH + cs;
      v8h k0 = ((const v8h*)kp)[0], k1 = ((const v8h*)kp)[1],
          k2 = ((const v8h*)kp)[2], k3 = ((const v8h*)kp)[3];
      *(v8h*)&ks[r * 72 + cs]      = k0;
      *(v8h*)&ks[r * 72 + cs + 8]  = k1;
      *(v8h*)&ks[r * 72 + cs + 16] = k2;
      *(v8h*)&ks[r * 72 + cs + 24] = k3;
      v8h v0 = ((const v8h*)vp)[0], v1 = ((const v8h*)vp)[1],
          v2 = ((const v8h*)vp)[2], v3 = ((const v8h*)vp)[3];
#pragma unroll
      for (int i = 0; i < 8; ++i) {
        vts[(cs + i)      * 136 + r] = v0[i];
        vts[(cs + 8 + i)  * 136 + r] = v1[i];
        vts[(cs + 16 + i) * 136 + r] = v2[i];
        vts[(cs + 24 + i) * 136 + r] = v3[i];
      }
      if (kt < qt) {  // prefetch next K tile row (global_prefetch_b8)
        __builtin_prefetch(kf + (srow + 128) * HH + cs, 0, 0);
        __builtin_prefetch(vf + (srow + 128) * HH + cs, 0, 0);
      }
    }
    __syncthreads();

    // ---- S = Q K^T  (8 column tiles of 16) ----
    v8f s[8];
#pragma unroll
    for (int nt = 0; nt < 8; ++nt) {
      const _Float16* kc = &ks[(nt * 16 + ln) * 72];
      v8f sa = vzero;
      sa = wmma_f16(qa0, ld_b(kc, hi), sa);
      sa = wmma_f16(qa1, ld_b(kc + 32, hi), sa);
      s[nt] = sa;
    }

    // ---- scale (+ causal mask on diagonal tile) ----
    if (kt == qt) {
#pragma unroll
      for (int nt = 0; nt < 8; ++nt)
#pragma unroll
        for (int j = 0; j < 8; ++j) {
          const int col = kt * 128 + nt * 16 + ln;
          const int row = qrow0 + j + hi * 8;
          s[nt][j] = (col <= row) ? s[nt][j] * scale : -1e30f;
        }
    } else {
#pragma unroll
      for (int nt = 0; nt < 8; ++nt)
#pragma unroll
        for (int j = 0; j < 8; ++j) s[nt][j] *= scale;
    }

    // ---- online softmax: row max / rescale / exp / row sum ----
    float mn[8];
#pragma unroll
    for (int j = 0; j < 8; ++j) {
      float v = s[0][j];
#pragma unroll
      for (int nt = 1; nt < 8; ++nt) v = fmaxf(v, s[nt][j]);
      v = fmaxf(v, __shfl_xor(v, 1, 32));
      v = fmaxf(v, __shfl_xor(v, 2, 32));
      v = fmaxf(v, __shfl_xor(v, 4, 32));
      v = fmaxf(v, __shfl_xor(v, 8, 32));   // reduce within 16-lane half
      mn[j] = fmaxf(m[j], v);
    }
    float f[8];
#pragma unroll
    for (int j = 0; j < 8; ++j) {
      f[j] = __expf(m[j] - mn[j]);
      m[j] = mn[j];
      l[j] *= f[j];
    }
#pragma unroll
    for (int nt = 0; nt < 4; ++nt)
#pragma unroll
      for (int j = 0; j < 8; ++j) o[nt][j] *= f[j];

    float rs[8];
#pragma unroll
    for (int j = 0; j < 8; ++j) rs[j] = 0.f;
#pragma unroll
    for (int nt = 0; nt < 8; ++nt)
#pragma unroll
      for (int j = 0; j < 8; ++j) {
        const float p = __expf(s[nt][j] - m[j]);
        s[nt][j] = p;
        rs[j] += p;
      }
#pragma unroll
    for (int j = 0; j < 8; ++j) {
      float v = rs[j];
      v += __shfl_xor(v, 1, 32);
      v += __shfl_xor(v, 2, 32);
      v += __shfl_xor(v, 4, 32);
      v += __shfl_xor(v, 8, 32);
      l[j] += v;
    }

    // ---- O += P @ V : 4 K-chunks of 32, P via per-wave LDS staging ----
    _Float16* st = &stage[wv_ * 16 * 40];
#pragma unroll
    for (int pair = 0; pair < 4; ++pair) {
      const int nt0 = pair * 2;
#pragma unroll
      for (int d = 0; d < 2; ++d)
#pragma unroll
        for (int j = 0; j < 8; ++j)
          st[(j + hi * 8) * 40 + d * 16 + ln] = (_Float16)s[nt0 + d][j];
      // same-wave DS store->load ordering; keep the compiler honest too
      asm volatile("s_wait_dscnt 0x0" ::: "memory");
      const v16h pa = ld_a(&st[ln * 40], hi);
#pragma unroll
      for (int nt = 0; nt < 4; ++nt)
        o[nt] = wmma_f16(pa, ld_b(&vts[(nt * 16 + ln) * 136 + pair * 32], hi),
                         o[nt]);
    }
    __syncthreads();   // protect ks/vts for next key tile
  }

  // ---- epilogue: O / l -> out (fp32) ----
  float invl[8];
#pragma unroll
  for (int j = 0; j < 8; ++j) invl[j] = 1.f / l[j];
#pragma unroll
  for (int nt = 0; nt < 4; ++nt)
#pragma unroll
    for (int j = 0; j < 8; ++j)
      out[((size_t)b * TT + qrow0 + j + hi * 8) * HH + nt * 16 + ln] =
          o[nt][j] * invl[j];
}

// ---------------------------------------------------------------------------
extern "C" void kernel_launch(void* const* d_in, const int* in_sizes, int n_in,
                              void* d_out, int out_size, void* d_ws, size_t ws_size,
                              hipStream_t stream) {
  const float* x  = (const float*)d_in[0];
  const float* wk = (const float*)d_in[1];
  const float* wq = (const float*)d_in[2];
  const float* wv = (const float*)d_in[3];
  float* out = (float*)d_out;

  // workspace layout (f16): wT[3][64][1024] | q | k | v  (each [B*T][64])
  _Float16* ws = (_Float16*)d_ws;
  _Float16* wT = ws;
  _Float16* qf = wT + (size_t)3 * HH * CC;
  _Float16* kf = qf + (size_t)BB * TT * HH;
  _Float16* vf = kf + (size_t)BB * TT * HH;

  wt_kernel<<<dim3(256, 3), 256, 0, stream>>>(wq, wk, wv, wT);
  proj_kernel<<<dim3((BB * TT) / 128, 3), 256, 0, stream>>>(x, wT, qf, kf, vf);
  attn_kernel<<<dim3(TT / 128, BB), 256, 0, stream>>>(qf, kf, vf, out);
}
